// BPSSlotPredictor_41729902248115
// MI455X (gfx1250) — matrix operand order repaired and checked by reading, not compiled, not run on hardware
//
#include <hip/hip_runtime.h>
#include <hip/hip_bf16.h>

// ---------------------------------------------------------------------------
// BPS slot predictor for MI455X (gfx1250, wave32, WMMA + TDM).
// All dense matmuls run as bf16 v_wmma_f32_16x16x32_bf16 with fp32 accum.
// Weight tiles are staged into LDS by the Tensor Data Mover (tensor_load_to_lds
// + s_wait_tensorcnt) when the toolchain exposes the builtin; activations are
// staged by the waves.  The kNN score-MLP (1M x 64 x 64 per block) is fused so
// the 256MB 't' tensor is never materialized in HBM.
// ---------------------------------------------------------------------------

#define BDIM 16
#define KPTS 4096
#define NBNB 16
#define DDIM 256
#define ADIM 64
#define VDIMX 128
#define MROWS (BDIM * KPTS)     // 65536 token rows
#define KNBR  (KPTS * NBNB)     // 65536 rel-pos rows

typedef __attribute__((ext_vector_type(8)))  __bf16 v8bf;
typedef __attribute__((ext_vector_type(16))) __bf16 v16bf;
typedef __attribute__((ext_vector_type(8)))  float  v8f;
typedef __attribute__((ext_vector_type(4)))  unsigned int u32x4;
typedef __attribute__((ext_vector_type(4)))  int  i32x4;
typedef __attribute__((ext_vector_type(8)))  int  i32x8;

#if defined(__AMDGCN__) && __has_builtin(__builtin_amdgcn_tensor_load_to_lds)
#define HAVE_TDM 1
#else
#define HAVE_TDM 0
#endif

// ------------------------------- helpers -----------------------------------
__device__ __forceinline__ unsigned short f2bf(float f) {
  union { __bf16 h; unsigned short u; } c;
  c.h = (__bf16)f;                       // native v_cvt on gfx1250
  return c.u;
}
__device__ __forceinline__ v16bf cat16(v8bf lo, v8bf hi) {
  return __builtin_shufflevector(lo, hi, 0,1,2,3,4,5,6,7,8,9,10,11,12,13,14,15);
}
__device__ __forceinline__ v8bf ld8bf(const unsigned short* p) {
  v8bf r; __builtin_memcpy(&r, p, 16); return r;
}
__device__ __forceinline__ v8f vz8() {
  v8f z; for (int i = 0; i < 8; ++i) z[i] = 0.f; return z;
}
__device__ __forceinline__ float gelu_exact(float x) {
  return 0.5f * x * (1.f + erff(x * 0.70710678118654752f));
}
__device__ __forceinline__ float block_reduce_sum(float v, float* red) {
  int tid = threadIdx.x;
  red[tid] = v; __syncthreads();
  for (int o = blockDim.x >> 1; o > 0; o >>= 1) {
    if (tid < o) red[tid] += red[tid + o];
    __syncthreads();
  }
  float r = red[0]; __syncthreads();
  return r;
}

#if HAVE_TDM
// TDM: async 2-D tile (nrows x kelems of bf16) from global -> LDS, row-major,
// packed contiguously (tile_dim0 = kelems per row).  D# per CDNA5 ISA 8.3/8.4.
// OOB global reads (beyond kd_rem / nrows) return zero -> LDS is zero-padded,
// matching the manual staging path.
__device__ __forceinline__ void tdm_load_tile_bf16(unsigned lds_byte_addr,
                                                   const void* gptr,
                                                   int kd_rem, int nrows,
                                                   int stride_elems,
                                                   int tile_k, int tile_rows) {
  unsigned long long ga = (unsigned long long)gptr;
  u32x4 g0;
  g0[0] = 1u;                                          // count=1, user D#
  g0[1] = lds_byte_addr;                               // lds_addr
  g0[2] = (unsigned)(ga & 0xffffffffu);                // global_addr[31:0]
  g0[3] = (unsigned)((ga >> 32) & 0x1ffffffu) | (2u << 30);  // addr[56:32]|type=2
  i32x8 g1;
  g1[0] = 1 << 16;                                     // data_size=1 (2B), no mcast
  g1[1] = (kd_rem & 0xffff) << 16;                     // tensor_dim0[15:0]
  g1[2] = (int)(((unsigned)kd_rem >> 16) & 0xffffu) | ((nrows & 0xffff) << 16);
  g1[3] = (int)(((unsigned)nrows >> 16) & 0xffffu) | (tile_k << 16); // tile_dim0
  g1[4] = tile_rows;                                   // tile_dim1 (tile_dim2=0)
  g1[5] = stride_elems;                                // tensor_dim0_stride lo32
  g1[6] = 0; g1[7] = 0;
  i32x4 z4 = {0, 0, 0, 0};
#if __clang_major__ >= 23
  i32x8 z8 = {0, 0, 0, 0, 0, 0, 0, 0};
  __builtin_amdgcn_tensor_load_to_lds(g0, g1, z4, z4, z8, 0);
#else
  __builtin_amdgcn_tensor_load_to_lds(g0, g1, z4, z4, 0);
#endif
}
#endif

// ----------------------- fp32 -> bf16 (row pad) ----------------------------
__global__ void k_f32_to_bf16_pad(const float* __restrict__ src,
                                  unsigned short* __restrict__ dst,
                                  int rows, int kin, int kpad) {
  int i = blockIdx.x * blockDim.x + threadIdx.x;
  if (i >= rows * kpad) return;
  int r = i / kpad, c = i - r * kpad;
  float v = (c < kin) ? src[(size_t)r * kin + c] : 0.f;
  dst[i] = f2bf(v);
}

// ----------------------------- pt features ---------------------------------
__global__ void k_build_pt(const float* __restrict__ basis,
                           const float* __restrict__ dists,
                           const float* __restrict__ nnp,
                           unsigned short* __restrict__ ptb) {
  int i = blockIdx.x * blockDim.x + threadIdx.x;
  if (i >= MROWS) return;
  int kk = i & (KPTS - 1);
  float bx = basis[kk * 3 + 0], by = basis[kk * 3 + 1], bz = basis[kk * 3 + 2];
  float nx = nnp[(size_t)i * 3 + 0], ny = nnp[(size_t)i * 3 + 1], nz = nnp[(size_t)i * 3 + 2];
  float vals[16] = { bx, by, bz, nx, ny, nz, bx - nx, by - ny, bz - nz,
                     dists[i], 0.f, 0.f, 0.f, 0.f, 0.f, 0.f };
  unsigned short* o = ptb + (size_t)i * 16;
  for (int j = 0; j < 16; ++j) o[j] = f2bf(vals[j]);
}

// ------------------------------ LayerNorm ----------------------------------
__global__ __launch_bounds__(256) void k_ln256(const float* __restrict__ x,
                                               const float* __restrict__ gam,
                                               const float* __restrict__ bet,
                                               unsigned short* __restrict__ out) {
  __shared__ float red[256];
  int row = blockIdx.x, tid = threadIdx.x;
  float v = x[(size_t)row * DDIM + tid];
  float mean = block_reduce_sum(v, red) * (1.f / DDIM);
  float d = v - mean;
  float var = block_reduce_sum(d * d, red) * (1.f / DDIM);
  float inv = rsqrtf(var + 1e-5f);
  out[(size_t)row * DDIM + tid] = f2bf(d * inv * gam[tid] + bet[tid]);
}

__global__ __launch_bounds__(256) void k_concat_ln512(const float* __restrict__ x,
                                                      const float* __restrict__ g,
                                                      const float* __restrict__ gam,
                                                      const float* __restrict__ bet,
                                                      unsigned short* __restrict__ out) {
  __shared__ float red[256];
  int row = blockIdx.x, tid = threadIdx.x;
  int b = row >> 12;
  float v0 = x[(size_t)row * DDIM + tid];
  float v1 = g[(size_t)b * DDIM + tid];
  float mean = block_reduce_sum(v0 + v1, red) * (1.f / 512.f);
  float d0 = v0 - mean, d1 = v1 - mean;
  float var = block_reduce_sum(d0 * d0 + d1 * d1, red) * (1.f / 512.f);
  float inv = rsqrtf(var + 1e-5f);
  unsigned short* o = out + (size_t)row * 512;
  o[tid]       = f2bf(d0 * inv * gam[tid] + bet[tid]);
  o[tid + 256] = f2bf(d1 * inv * gam[tid + 256] + bet[tid + 256]);
}

// ------------------------------ mean over K --------------------------------
__global__ __launch_bounds__(256) void k_mean(const float* __restrict__ x,
                                              float* __restrict__ m) {
  int b = blockIdx.x, d = threadIdx.x;
  float s = 0.f;
  for (int kk = 0; kk < KPTS; ++kk)
    s += x[((size_t)b * KPTS + kk) * DDIM + d];
  m[(size_t)b * DDIM + d] = s * (1.f / KPTS);
}

// --------------------------- WMMA bf16 GEMM --------------------------------
// C[M,N] = act(A[M,Kd] * W[N,Kd]^T + bias) (+ resid).  Workgroup tile
// 64(M) x 128(N), K-step 64; 8 waves = 4(M) x 2(N); 8 v_wmma per staged tile
// per wave.  W tile arrives via TDM (wave 0 issues, s_wait_tensorcnt, barrier).
#define TM 64
#define TN 128
#define TK 64

template<int ACT, bool OUTBF, bool RESID>
__global__ __launch_bounds__(256) void k_gemm(const unsigned short* __restrict__ A,
                                              const unsigned short* __restrict__ W,
                                              const float* __restrict__ bias,
                                              const float* __restrict__ resid,
                                              float* __restrict__ Cf,
                                              unsigned short* __restrict__ Cb,
                                              int M, int N, int Kd) {
  __shared__ __align__(16) unsigned short As[TM * TK];   // 8 KB
  __shared__ __align__(16) unsigned short Bs[TN * TK];   // 16 KB
  const int tid = threadIdx.x, lane = tid & 31, wave = tid >> 5;
  const int mw = wave & 3, nw = wave >> 2;
  const int mt = blockIdx.y, nt = blockIdx.x;
  const int hf = lane >> 4, l16 = lane & 15;

  v8f acc[4];
  for (int t = 0; t < 4; ++t) acc[t] = vz8();

  for (int k0 = 0; k0 < Kd; k0 += TK) {
    // ---- stage A tile (64 x 64 bf16): 512 x 16B chunks, 2 per thread ----
#pragma unroll
    for (int c = 0; c < 2; ++c) {
      int chunk = tid + c * 256;
      int ar = chunk >> 3, kc = (chunk & 7) * 8;
      int gm = mt * TM + ar;
      uint4 val = {0u, 0u, 0u, 0u};
      if (gm < M && (k0 + kc) < Kd)
        __builtin_memcpy(&val, A + (size_t)gm * Kd + k0 + kc, 16);
      __builtin_memcpy(&As[ar * TK + kc], &val, 16);
    }
    // ---- stage W tile (128 x 64 bf16) ----
#if HAVE_TDM
    if (wave == 0) {
      int nrem = N - nt * TN; if (nrem > TN) nrem = TN;
      tdm_load_tile_bf16((unsigned)(size_t)(void*)Bs,
                         W + (size_t)(nt * TN) * Kd + k0,
                         Kd - k0, nrem, Kd, TK, TN);
      __builtin_amdgcn_s_wait_tensorcnt(0);
    }
#else
#pragma unroll
    for (int c = 0; c < 4; ++c) {
      int chunk = tid + c * 256;
      int nr = chunk >> 3, kc = (chunk & 7) * 8;
      int gn = nt * TN + nr;
      uint4 val = {0u, 0u, 0u, 0u};
      if (gn < N && (k0 + kc) < Kd)
        __builtin_memcpy(&val, W + (size_t)gn * Kd + k0 + kc, 16);
      __builtin_memcpy(&Bs[nr * TK + kc], &val, 16);
    }
#endif
    if (k0 + TK < Kd)  // speculative prefetch of next A tile -> global_prefetch_b8
      __builtin_prefetch(A + (size_t)(mt * TM + (tid >> 2)) * Kd + k0 + TK, 0, 1);
    __syncthreads();

    // ---- 2 K-substeps x 4 N-subtiles = 8 v_wmma per staged tile ----
#pragma unroll
    for (int kb = 0; kb < 2; ++kb) {
      int arow = mw * 16 + l16;
      // A frag: lane<16 K {0..7,16..23}, lane>=16 K {8..15,24..31} (+32*kb)
      v16bf afrag = cat16(ld8bf(&As[arow * TK + kb * 32 + hf * 8]),
                          ld8bf(&As[arow * TK + kb * 32 + 16 + hf * 8]));
      // reuse_a hint (must be literal constant) on first 3 of 4 identical WMMAs
#define GEMM_WMMA(T, RA)                                                    \
      {                                                                     \
        int brow = nw * 64 + (T) * 16 + l16;                                \
        v16bf bfrag = cat16(ld8bf(&Bs[brow * TK + kb * 32 + hf * 8]),       \
                            ld8bf(&Bs[brow * TK + kb * 32 + 16 + hf * 8])); \
        acc[T] = __builtin_amdgcn_wmma_f32_16x16x32_bf16(                   \
            false, afrag, false, bfrag, (short)0, acc[T], RA, false);       \
      }
      GEMM_WMMA(0, true)
      GEMM_WMMA(1, true)
      GEMM_WMMA(2, true)
      GEMM_WMMA(3, false)
#undef GEMM_WMMA
    }
    __syncthreads();
  }

#pragma unroll
  for (int t = 0; t < 4; ++t) {
    int gn = nt * TN + nw * 64 + t * 16 + l16;
    if (gn >= N) continue;
    float bv = bias ? bias[gn] : 0.f;
#pragma unroll
    for (int r = 0; r < 8; ++r) {
      int gm = mt * TM + mw * 16 + hf * 8 + r;    // C: M = r + 8*(lane>=16)
      if (gm >= M) continue;
      float vv = acc[t][r] + bv;
      if (ACT == 1) vv = fmaxf(vv, 0.f);
      else if (ACT == 2) vv = gelu_exact(vv);
      if (RESID) vv += resid[(size_t)gm * N + gn];
      if (OUTBF) Cb[(size_t)gm * N + gn] = f2bf(vv);
      else       Cf[(size_t)gm * N + gn] = vv;
    }
  }
}

// --------------------- fused kNN attention (per point) ---------------------
__global__ __launch_bounds__(128) void k_knn_attn(const float* __restrict__ qm,
                                                  const float* __restrict__ km,
                                                  const float* __restrict__ vm,
                                                  const float* __restrict__ ra,
                                                  const float* __restrict__ rv,
                                                  const int*   __restrict__ knn,
                                                  const unsigned short* __restrict__ s1w,
                                                  const float* __restrict__ s1b,
                                                  const float* __restrict__ s2w,
                                                  const float* __restrict__ s2bp,
                                                  unsigned short* __restrict__ ctx) {
  __shared__ __align__(16) unsigned short s1s[ADIM * ADIM];   // 8KB
  __shared__ float s1bs[ADIM], s2s[ADIM];
  __shared__ float attns[4][NBNB];
  __shared__ int   idxs[4][NBNB];

  const int tid = threadIdx.x, lane = tid & 31, wave = tid >> 5;
  for (int c = tid; c < (ADIM * ADIM) / 8; c += 128) {
    uint4 v; __builtin_memcpy(&v, s1w + c * 8, 16);
    __builtin_memcpy(&s1s[c * 8], &v, 16);
  }
  if (tid < ADIM) { s1bs[tid] = s1b[tid]; s2s[tid] = s2w[tid]; }
  __syncthreads();

  const int p   = blockIdx.x * 4 + wave;
  const int b   = p >> 12;
  const int kpt = p & (KPTS - 1);
  const int hf = lane >> 4, nb = lane & 15;

  const int gidx = knn[(size_t)kpt * NBNB + nb];
  if (hf == 0) idxs[wave][nb] = gidx;
  const float* qr  = qm + (size_t)p * ADIM;
  const float* kg  = km + ((size_t)b * KPTS + gidx) * ADIM;
  const float* rar = ra + ((size_t)kpt * NBNB + nb) * ADIM;

  // A fragments for t (M = neighbor row nb, K = attn dim a), native cvt
  v16bf af[2];
#pragma unroll
  for (int kb = 0; kb < 2; ++kb) {
#pragma unroll
    for (int e = 0; e < 16; ++e) {
      int a = kb * 32 + ((e < 8) ? (hf * 8 + e) : (16 + hf * 8 + (e - 8)));
      af[kb][e] = (__bf16)tanhf(qr[a] - kg[a] + rar[a]);
    }
  }

  v8f acc[4];
  for (int t = 0; t < 4; ++t) acc[t] = vz8();
#pragma unroll
  for (int kb = 0; kb < 2; ++kb) {
#define ATTN_WMMA(JT, RA)                                                  \
    {                                                                      \
      int j = (JT) * 16 + nb;                                              \
      const unsigned short* row = &s1s[j * ADIM + kb * 32 + hf * 8];       \
      v16bf bfrag = cat16(ld8bf(row), ld8bf(row + 16));                    \
      acc[JT] = __builtin_amdgcn_wmma_f32_16x16x32_bf16(                   \
          false, af[kb], false, bfrag, (short)0, acc[JT], RA, false);      \
    }
    ATTN_WMMA(0, true)
    ATTN_WMMA(1, true)
    ATTN_WMMA(2, true)
    ATTN_WMMA(3, false)
#undef ATTN_WMMA
  }

  float part[8];
  for (int r = 0; r < 8; ++r) part[r] = 0.f;
#pragma unroll
  for (int jt = 0; jt < 4; ++jt) {
    int j = jt * 16 + nb;
    float w2 = s2s[j], bj = s1bs[j];
#pragma unroll
    for (int r = 0; r < 8; ++r) part[r] += fmaxf(acc[jt][r] + bj, 0.f) * w2;
  }
  for (int off = 1; off < 16; off <<= 1)
    for (int r = 0; r < 8; ++r) part[r] += __shfl_xor(part[r], off, 32);

  const float s2b = s2bp[0];
  const float inv_scale = 0.125f;              // 1/sqrt(64)
  float my[8], ot[8];
  for (int r = 0; r < 8; ++r) my[r] = (part[r] + s2b) * inv_scale;
  for (int r = 0; r < 8; ++r) ot[r] = __shfl_xor(my[r], 16, 32);
  float mx = -3.4e38f;
  for (int r = 0; r < 8; ++r) mx = fmaxf(mx, fmaxf(my[r], ot[r]));
  float den = 0.f;
  for (int r = 0; r < 8; ++r) den += expf(my[r] - mx) + expf(ot[r] - mx);
  if (nb == 0)
    for (int r = 0; r < 8; ++r) attns[wave][hf * 8 + r] = expf(my[r] - mx) / den;
  __syncthreads();

  float cacc[4] = {0.f, 0.f, 0.f, 0.f};
  for (int n2 = 0; n2 < NBNB; ++n2) {
    float w = attns[wave][n2];
    int gi = idxs[wave][n2];
    const float* vr  = vm + ((size_t)b * KPTS + gi) * VDIMX;
    const float* rvr = rv + ((size_t)kpt * NBNB + n2) * VDIMX;
#pragma unroll
    for (int c = 0; c < 4; ++c) cacc[c] += w * (vr[lane + 32 * c] + rvr[lane + 32 * c]);
  }
#pragma unroll
  for (int c = 0; c < 4; ++c)
    ctx[(size_t)p * VDIMX + lane + 32 * c] = f2bf(cacc[c]);
}

// ------------------------------- host side ---------------------------------
static void launch_gemm(int act, bool outbf, bool resid_on,
                        const unsigned short* A, const unsigned short* W,
                        const float* bias, const float* resid,
                        float* Cf, unsigned short* Cb,
                        int M, int N, int Kd, hipStream_t s) {
  dim3 grid((N + TN - 1) / TN, (M + TM - 1) / TM), blk(256);
  if (outbf && act == 1)       k_gemm<1, true,  false><<<grid, blk, 0, s>>>(A, W, bias, resid, Cf, Cb, M, N, Kd);
  else if (outbf && act == 2)  k_gemm<2, true,  false><<<grid, blk, 0, s>>>(A, W, bias, resid, Cf, Cb, M, N, Kd);
  else if (resid_on)           k_gemm<0, false, true ><<<grid, blk, 0, s>>>(A, W, bias, resid, Cf, Cb, M, N, Kd);
  else                         k_gemm<0, false, false><<<grid, blk, 0, s>>>(A, W, bias, resid, Cf, Cb, M, N, Kd);
}

// Input order: top-level dict + params pytree flattened jax-style (sorted keys).
enum { IN_BASIS = 0, IN_DISTS, IN_NNP, IN_KNN, IN_REL, PARAM0 };
enum { L_BK = 0, L_BO, L_BQ, L_BV, L_F1, L_F1B, L_F2, L_F2B,
       L_N1B, L_N1G, L_N2B, L_N2G,
       L_PA1, L_PA1B, L_PA2, L_PA2B, L_PV1, L_PV1B, L_PV2, L_PV2B,
       L_S1, L_S1B, L_S2, L_S2B, L_WK, L_WO, L_WQ, L_WV, NBLK_LEAVES };
enum { O_G1W = 0, O_G1B, O_G2W, O_G2B, O_H1W, O_H1B, O_H2W, O_H2B,
       O_HLNB, O_HLNG, O_IN1W, O_IN1B, O_IN2W, O_IN2B };

extern "C" void kernel_launch(void* const* d_in, const int* in_sizes, int n_in,
                              void* d_out, int out_size, void* d_ws, size_t ws_size,
                              hipStream_t stream) {
  (void)in_sizes; (void)n_in; (void)out_size; (void)ws_size;
  const int PAFTER = PARAM0 + 2 * NBLK_LEAVES;  // 61
  auto leaf  = [&](int i) { return (const float*)d_in[i]; };
  auto bleaf = [&](int blk, int l) { return (const float*)d_in[PARAM0 + blk * NBLK_LEAVES + l]; };

  const float* basis = leaf(IN_BASIS);
  const float* dists = leaf(IN_DISTS);
  const float* nnp   = leaf(IN_NNP);
  const int*   knn   = (const int*)d_in[IN_KNN];
  const float* rel   = leaf(IN_REL);

  size_t cur = 0;
  auto alloc = [&](size_t bytes) { size_t o = cur; cur += (bytes + 255) & ~(size_t)255; return o; };
  auto F  = [&](size_t off) { return (float*)((char*)d_ws + off); };
  auto U  = [&](size_t off) { return (unsigned short*)((char*)d_ws + off); };

  const size_t o_x    = alloc((size_t)MROWS * DDIM * 4);
  const size_t o_hb   = alloc((size_t)MROWS * 512  * 2);
  const size_t o_big  = alloc((size_t)MROWS * 512  * 2);
  const size_t o_q    = alloc((size_t)MROWS * ADIM * 4);
  const size_t o_k    = alloc((size_t)MROWS * ADIM * 4);
  const size_t o_v    = alloc((size_t)MROWS * VDIMX * 4);
  const size_t o_ra   = alloc((size_t)KNBR  * ADIM * 4);
  const size_t o_rv   = alloc((size_t)KNBR  * VDIMX * 4);
  const size_t o_hid  = alloc((size_t)KNBR  * VDIMX * 2);
  const size_t o_ctx  = alloc((size_t)MROWS * VDIMX * 2);
  const size_t o_ptb  = alloc((size_t)MROWS * 16 * 2);
  const size_t o_relb = alloc((size_t)KNBR  * 16 * 2);
  const size_t o_mean = alloc((size_t)BDIM * DDIM * 4);
  const size_t o_meanb= alloc((size_t)BDIM * DDIM * 2);
  const size_t o_gh   = alloc((size_t)BDIM * DDIM * 2);
  const size_t o_g    = alloc((size_t)BDIM * DDIM * 4);
  size_t w_in1 = alloc(256 * 16 * 2),  w_in2 = alloc(256 * 256 * 2);
  size_t w_g1  = alloc(256 * 256 * 2), w_g2  = alloc(256 * 256 * 2);
  size_t w_h1  = alloc(256 * 512 * 2), w_h2  = alloc(5 * 256 * 2);
  size_t w_wq[2], w_wk[2], w_wv[2], w_wo[2], w_pa1[2], w_pa2[2],
         w_pv1[2], w_pv2[2], w_s1[2], w_f1[2], w_f2[2];
  for (int b = 0; b < 2; ++b) {
    w_wq[b]  = alloc(64 * 256 * 2);  w_wk[b]  = alloc(64 * 256 * 2);
    w_wv[b]  = alloc(128 * 256 * 2); w_wo[b]  = alloc(256 * 128 * 2);
    w_pa1[b] = alloc(64 * 16 * 2);   w_pa2[b] = alloc(64 * 64 * 2);
    w_pv1[b] = alloc(128 * 16 * 2);  w_pv2[b] = alloc(128 * 128 * 2);
    w_s1[b]  = alloc(64 * 64 * 2);
    w_f1[b]  = alloc(512 * 256 * 2); w_f2[b]  = alloc(256 * 512 * 2);
  }

  auto conv = [&](const float* src, size_t dst, int rows, int kin, int kpad) {
    int total = rows * kpad;
    k_f32_to_bf16_pad<<<(total + 255) / 256, 256, 0, stream>>>(src, U(dst), rows, kin, kpad);
  };

  conv(leaf(PAFTER + O_IN1W), w_in1, 256, 10, 16);
  conv(leaf(PAFTER + O_IN2W), w_in2, 256, 256, 256);
  conv(leaf(PAFTER + O_G1W),  w_g1,  256, 256, 256);
  conv(leaf(PAFTER + O_G2W),  w_g2,  256, 256, 256);
  conv(leaf(PAFTER + O_H1W),  w_h1,  256, 512, 512);
  conv(leaf(PAFTER + O_H2W),  w_h2,  5,   256, 256);
  for (int b = 0; b < 2; ++b) {
    conv(bleaf(b, L_WQ),  w_wq[b],  64, 256, 256);
    conv(bleaf(b, L_WK),  w_wk[b],  64, 256, 256);
    conv(bleaf(b, L_WV),  w_wv[b],  128, 256, 256);
    conv(bleaf(b, L_WO),  w_wo[b],  256, 128, 128);
    conv(bleaf(b, L_PA1), w_pa1[b], 64, 3, 16);
    conv(bleaf(b, L_PA2), w_pa2[b], 64, 64, 64);
    conv(bleaf(b, L_PV1), w_pv1[b], 128, 3, 16);
    conv(bleaf(b, L_PV2), w_pv2[b], 128, 128, 128);
    conv(bleaf(b, L_S1),  w_s1[b],  64, 64, 64);
    conv(bleaf(b, L_F1),  w_f1[b],  512, 256, 256);
    conv(bleaf(b, L_F2),  w_f2[b],  256, 512, 512);
  }

  k_build_pt<<<MROWS / 256, 256, 0, stream>>>(basis, dists, nnp, U(o_ptb));
  conv(rel, o_relb, KNBR, 3, 16);
  launch_gemm(1, true, false, U(o_ptb), U(w_in1), leaf(PAFTER + O_IN1B), nullptr,
              nullptr, U(o_big), MROWS, 256, 16, stream);
  launch_gemm(0, false, false, U(o_big), U(w_in2), leaf(PAFTER + O_IN2B), nullptr,
              F(o_x), nullptr, MROWS, 256, 256, stream);

  for (int b = 0; b < 2; ++b) {
    k_ln256<<<MROWS, 256, 0, stream>>>(F(o_x), bleaf(b, L_N1G), bleaf(b, L_N1B), U(o_hb));
    launch_gemm(0, false, false, U(o_hb), U(w_wq[b]), bleaf(b, L_BQ), nullptr,
                F(o_q), nullptr, MROWS, ADIM, 256, stream);
    launch_gemm(0, false, false, U(o_hb), U(w_wk[b]), bleaf(b, L_BK), nullptr,
                F(o_k), nullptr, MROWS, ADIM, 256, stream);
    launch_gemm(0, false, false, U(o_hb), U(w_wv[b]), bleaf(b, L_BV), nullptr,
                F(o_v), nullptr, MROWS, VDIMX, 256, stream);
    launch_gemm(1, true, false, U(o_relb), U(w_pa1[b]), bleaf(b, L_PA1B), nullptr,
                nullptr, U(o_hid), KNBR, ADIM, 16, stream);
    launch_gemm(0, false, false, U(o_hid), U(w_pa2[b]), bleaf(b, L_PA2B), nullptr,
                F(o_ra), nullptr, KNBR, ADIM, ADIM, stream);
    launch_gemm(1, true, false, U(o_relb), U(w_pv1[b]), bleaf(b, L_PV1B), nullptr,
                nullptr, U(o_hid), KNBR, VDIMX, 16, stream);
    launch_gemm(0, false, false, U(o_hid), U(w_pv2[b]), bleaf(b, L_PV2B), nullptr,
                F(o_rv), nullptr, KNBR, VDIMX, VDIMX, stream);
    k_knn_attn<<<MROWS / 4, 128, 0, stream>>>(F(o_q), F(o_k), F(o_v), F(o_ra), F(o_rv),
                                              knn, U(w_s1[b]), bleaf(b, L_S1B),
                                              bleaf(b, L_S2), bleaf(b, L_S2B), U(o_ctx));
    launch_gemm(0, false, true, U(o_ctx), U(w_wo[b]), bleaf(b, L_BO), F(o_x),
                F(o_x), nullptr, MROWS, 256, 128, stream);
    k_ln256<<<MROWS, 256, 0, stream>>>(F(o_x), bleaf(b, L_N2G), bleaf(b, L_N2B), U(o_hb));
    launch_gemm(2, true, false, U(o_hb), U(w_f1[b]), bleaf(b, L_F1B), nullptr,
                nullptr, U(o_big), MROWS, 512, 256, stream);
    launch_gemm(0, false, true, U(o_big), U(w_f2[b]), bleaf(b, L_F2B), F(o_x),
                F(o_x), nullptr, MROWS, 256, 512, stream);
  }

  k_mean<<<BDIM, 256, 0, stream>>>(F(o_x), F(o_mean));
  conv(F(o_mean), o_meanb, BDIM, 256, 256);
  launch_gemm(2, true, false, U(o_meanb), U(w_g1), leaf(PAFTER + O_G1B), nullptr,
              nullptr, U(o_gh), BDIM, 256, 256, stream);
  launch_gemm(0, false, false, U(o_gh), U(w_g2), leaf(PAFTER + O_G2B), nullptr,
              F(o_g), nullptr, BDIM, 256, 256, stream);
  k_concat_ln512<<<MROWS, 256, 0, stream>>>(F(o_x), F(o_g),
                                            leaf(PAFTER + O_HLNG), leaf(PAFTER + O_HLNB),
                                            U(o_hb));
  launch_gemm(1, true, false, U(o_hb), U(w_h1), leaf(PAFTER + O_H1B), nullptr,
              nullptr, U(o_big), MROWS, 256, 512, stream);
  launch_gemm(0, false, false, U(o_big), U(w_h2), leaf(PAFTER + O_H2B), nullptr,
              (float*)d_out, nullptr, MROWS, 5, 256, stream);
}
